// SemlaLayer_80161269612633
// MI455X (gfx1250) — compile-verified, matrix operand order checked by reading, not run on hardware
//
#include <hip/hip_runtime.h>
#include <stdint.h>

// ---------------------------------------------------------------------------
// SemlaLayer fused kernel for gfx1250 (MI455X).
// Flash-softmax fused pair-MLP using v_wmma_f32_16x16x32_bf16 +
// double-buffered async global->LDS staging (ASYNCcnt pipelined).
//   B=8, N=256, d_equi=64, d_inv=256, d_msg=64, H=8, d_ff=256
//   pair feats = 192 (qm|km|dot), msg channels = 72 (pad->80)
// Grid: 256 blocks = (b, 8-query tile); 8 waves/block; wave w owns query w.
// ---------------------------------------------------------------------------

#if __has_builtin(__builtin_amdgcn_global_load_async_to_lds_b128)
#define HAVE_ASYNC 1
#else
#define HAVE_ASYNC 0
#endif

// ---- workspace layout (bytes), total 4333568 B ----
#define WS_W1   0u
#define WS_W2   98304u
#define WS_QM   139264u
#define WS_KM   401408u
#define WS_PE   663552u
#define WS_PI   2236416u

// ---- LDS layout (bytes), all 16B aligned ----
#define L_W1   0u        // 24576 u32 = 98304
#define L_W2   98304u    // 10240 u32 = 40960
#define L_QM   139264u   // 8*64 u16  = 1024
#define L_EQQ  140288u   // 8*192 f32 = 6144
#define L_B1   146432u   // 256 f32   = 1024
#define L_B2   147456u   // 80 f32 (pad 512)
#define L_HID  147968u   // 8 waves * 16*256 bf16 = 65536
#define L_TILE 213504u   // 2 x 43520 double-buffered k-tile set
// within a tile set:
#define T_KM   0u        // 16*64 u16  = 2048
#define T_EQK  2048u     // 16*192 f32 = 12288
#define T_PE   14336u    // 16*192 f32 = 12288
#define T_PI   26624u    // 16*256 f32 = 16384
#define T_ADJ  43008u    // 8*16 i32   = 512
#define TILE_SZ 43520u
#define SMEM_BYTES (L_TILE + 2u * TILE_SZ)   // 300544 B < 320KB/WGP

#define NEGC (-10000000.0f)

typedef __attribute__((ext_vector_type(16))) __bf16 bf16x16;
typedef __attribute__((ext_vector_type(8)))  float  f32x8;
typedef __attribute__((vector_size(16)))     int    v4i_;

__device__ __forceinline__ uint16_t f2bf(float x) {
  uint32_t u = __float_as_uint(x);
  u += 0x7FFFu + ((u >> 16) & 1u);          // round-to-nearest-even
  return (uint16_t)(u >> 16);
}
__device__ __forceinline__ uint32_t pack2(uint16_t lo, uint16_t hi) {
  return (uint32_t)lo | ((uint32_t)hi << 16);
}

// -------------------------- prep kernels -----------------------------------

__global__ __launch_bounds__(256) void prep_nodes(
    const float* __restrict__ invs,
    const float* __restrict__ wq, const float* __restrict__ bq,
    const float* __restrict__ wk, const float* __restrict__ bk,
    const float* __restrict__ wii, const float* __restrict__ bii,
    uint16_t* __restrict__ qmb, uint16_t* __restrict__ kmb,
    float* __restrict__ proj_inv) {
  __shared__ float s[256];
  const int row = blockIdx.x, t = threadIdx.x;
  s[t] = invs[(size_t)row * 256 + t];
  __syncthreads();
  float a = bii[t];
  for (int j = 0; j < 256; ++j) a += s[j] * wii[j * 256 + t];
  proj_inv[(size_t)row * 256 + t] = a;
  if (t < 64) {
    float q = bq[t], k = bk[t];
    for (int j = 0; j < 256; ++j) {
      q += s[j] * wq[j * 64 + t];
      k += s[j] * wk[j * 64 + t];
    }
    qmb[(size_t)row * 64 + t] = f2bf(q);
    kmb[(size_t)row * 64 + t] = f2bf(k);
  }
}

__global__ __launch_bounds__(64) void prep_equi(
    const float* __restrict__ equis, const float* __restrict__ wco,
    float* __restrict__ pe) {
  __shared__ float s[64];
  const int row = blockIdx.x, t = threadIdx.x;
  s[t] = equis[(size_t)row * 64 + t];
  __syncthreads();
  float a = 0.f;
  for (int j = 0; j < 64; ++j) a += s[j] * wco[j * 64 + t];
  pe[(size_t)row * 64 + t] = a;
}

// pack w_mlp1 / w_mlp2 into per-lane WMMA bf16 B-fragment order
__global__ __launch_bounds__(256) void prep_pack(
    const float* __restrict__ w1, const float* __restrict__ w2,
    uint32_t* __restrict__ w1p, uint32_t* __restrict__ w2p) {
  const int idx = blockIdx.x * 256 + threadIdx.x;
  if (idx < 24576) {                       // w1: 16 ntiles * 6 kchunks * 32 * 8
    int j = idx / 1536;
    int rem = idx % 1536;
    int kc = rem / 256;
    int lane = (rem % 256) / 8;
    int v = rem & 7;
    int n = j * 16 + (lane & 15);
    int kbase = kc * 32 + ((lane >= 16) ? 8 : 0);
    int k0 = kbase + ((v < 4) ? 2 * v : 16 + 2 * (v - 4));
    w1p[idx] = pack2(f2bf(w1[k0 * 256 + n]), f2bf(w1[(k0 + 1) * 256 + n]));
  } else if (idx < 24576 + 10240) {        // w2: 5 ntiles * 8 kchunks * 32 * 8
    int i2 = idx - 24576;
    int j2 = i2 / 2048;
    int rem = i2 % 2048;
    int kc2 = rem / 256;
    int lane = (rem % 256) / 8;
    int v = rem & 7;
    int n = j2 * 16 + (lane & 15);
    int kbase = kc2 * 32 + ((lane >= 16) ? 8 : 0);
    int k0 = kbase + ((v < 4) ? 2 * v : 16 + 2 * (v - 4));
    uint16_t lo = (n < 72) ? f2bf(w2[k0 * 72 + n]) : (uint16_t)0;
    uint16_t hi = (n < 72) ? f2bf(w2[(k0 + 1) * 72 + n]) : (uint16_t)0;
    w2p[i2] = pack2(lo, hi);
  }
}

// pair feature as bf16: [qm(q) | km(k) | dot(q,k)]
__device__ __forceinline__ uint16_t getfeat(
    const uint16_t* __restrict__ QM, const uint16_t* __restrict__ KM,
    const float* __restrict__ EQQ, const float* __restrict__ EQK,
    int ql, int kl, int idx) {
  if (idx < 64) return QM[ql * 64 + idx];
  if (idx < 128) return KM[kl * 64 + (idx - 64)];
  const int d = idx - 128;
  float s = EQQ[ql * 192 + d]       * EQK[kl * 192 + d]
          + EQQ[ql * 192 + 64 + d]  * EQK[kl * 192 + 64 + d]
          + EQQ[ql * 192 + 128 + d] * EQK[kl * 192 + 128 + d];
  return f2bf(s);
}

// 16B copy: async DMA global->LDS if available, else plain copy.
// Builtin signature (from diagnostics): (v4i addrspace(1)*, v4i addrspace(3)*,
// imm offset, imm cpol).
#if HAVE_ASYNC
#define COPY16(l, g)                                                       \
  __builtin_amdgcn_global_load_async_to_lds_b128(                          \
      (__attribute__((address_space(1))) v4i_*)(g),                        \
      (__attribute__((address_space(3))) v4i_*)(l), 0, 0)
#else
#define COPY16(l, g) (*(uint4*)(l) = *(const uint4*)(g))
#endif

// Stage one 16-k tile (KM|EQK|PE|PI|ADJ = 2720 16B chunks, padded to 2816 so
// every wave issues exactly 11 async instructions -> wait threshold is exact).
__device__ __forceinline__ void issue_tile(
    char* tb, int tid, const float* __restrict__ equis,
    const int* __restrict__ adj, const uint16_t* __restrict__ km_bf,
    const float* __restrict__ proj_equi, const float* __restrict__ proj_inv,
    int b, int qt, int kb) {
  const char* gkm = (const char*)(km_bf + (size_t)(b * 256 + kb) * 64);
  const char* gek = (const char*)(equis + (size_t)(b * 256 + kb) * 192);
  const char* gpe = (const char*)(proj_equi + (size_t)(b * 256 + kb) * 192);
  const char* gpi = (const char*)(proj_inv + (size_t)(b * 256 + kb) * 256);
#pragma unroll
  for (int it = 0; it < 11; ++it) {
    const int i = tid + it * 256;
    const char* g;
    char* l;
    if (i < 128)        { g = gkm + i * 16;          l = tb + T_KM  + i * 16; }
    else if (i < 896)   { g = gek + (i - 128) * 16;  l = tb + T_EQK + (i - 128) * 16; }
    else if (i < 1664)  { g = gpe + (i - 896) * 16;  l = tb + T_PE  + (i - 896) * 16; }
    else if (i < 2688)  { g = gpi + (i - 1664) * 16; l = tb + T_PI  + (i - 1664) * 16; }
    else if (i < 2720)  {
      const int r = (i - 2688) >> 2, p = (i - 2688) & 3;
      g = (const char*)adj + (((size_t)(b * 256 + qt * 8 + r)) * 256 + kb) * 4 + p * 16;
      l = tb + T_ADJ + (i - 2688) * 16;
    } else              { g = gkm; l = tb + T_KM; }   // uniform-count pad
    COPY16(l, g);
  }
}

// -------------------------- main fused kernel ------------------------------

__global__ __launch_bounds__(256) void semla_main(
    const float* __restrict__ equis, const int* __restrict__ adj,
    const float* __restrict__ b_mlp1, const float* __restrict__ b_mlp2,
    const float* __restrict__ w_equi_out, const float* __restrict__ w_inv_out,
    const float* __restrict__ b_inv_out,
    const uint16_t* __restrict__ qm_bf, const uint16_t* __restrict__ km_bf,
    const float* __restrict__ proj_equi, const float* __restrict__ proj_inv,
    const uint32_t* __restrict__ w1p, const uint32_t* __restrict__ w2p,
    float* __restrict__ out) {
  extern __shared__ char smem[];
  uint32_t* W1  = (uint32_t*)(smem + L_W1);
  uint32_t* W2  = (uint32_t*)(smem + L_W2);
  uint16_t* QM  = (uint16_t*)(smem + L_QM);
  float*    EQQ = (float*)(smem + L_EQQ);
  float*    B1  = (float*)(smem + L_B1);
  float*    B2  = (float*)(smem + L_B2);
  uint16_t* HID = (uint16_t*)(smem + L_HID);

  const int tid  = threadIdx.x;
  const int wave = tid >> 5;
  const int lane = tid & 31;
  const int lhal = lane >> 4;
  const int lrow = lane & 15;
  const int bb = blockIdx.x;
  const int b  = bb >> 5;       // batch
  const int qt = bb & 31;       // 8-query tile index
  const int ql = wave;          // this wave's local query (0..7)

  // ---- one-time staging ----
  for (int i = tid; i < 24576; i += 256) W1[i] = w1p[i];
  for (int i = tid; i < 10240; i += 256) W2[i] = w2p[i];
  {
    const uint32_t* q32 = (const uint32_t*)(qm_bf + (size_t)(b * 256 + qt * 8) * 64);
    ((uint32_t*)QM)[tid] = q32[tid];   // 256 u32 = 8*64 bf16
    const float* eq = equis + (size_t)(b * 256 + qt * 8) * 192;
    for (int i = tid; i < 1536; i += 256) EQQ[i] = eq[i];
    B1[tid] = b_mlp1[tid];
    if (tid < 80) B2[tid] = (tid < 72) ? b_mlp2[tid] : 0.0f;
  }

  // ---- flash-softmax state (per lane; channel = group*16 + lrow) ----
  float m_[5], S_[5], S2_[5];
  float oe[4][3];
  float4 oi[8];
#pragma unroll
  for (int j = 0; j < 5; ++j) { m_[j] = -3.0e38f; S_[j] = 0.f; S2_[j] = 0.f; }
#pragma unroll
  for (int j = 0; j < 4; ++j)
    for (int c = 0; c < 3; ++c) oe[j][c] = 0.f;
#pragma unroll
  for (int d = 0; d < 8; ++d) oi[d] = make_float4(0.f, 0.f, 0.f, 0.f);

  uint16_t* hidw = HID + wave * 4096;
  const uint32_t* hid32 = (const uint32_t*)hidw;

  // prologue: issue async staging for tile 0
  issue_tile(smem + L_TILE, tid, equis, adj, km_bf, proj_equi, proj_inv,
             b, qt, 0);

  for (int kt = 0; kt < 16; ++kt) {
    char* tb = smem + L_TILE + (unsigned)(kt & 1) * TILE_SZ;
    __syncthreads();                       // all waves done with other buffer
    if (kt + 1 < 16)
      issue_tile(smem + L_TILE + (unsigned)((kt + 1) & 1) * TILE_SZ, tid,
                 equis, adj, km_bf, proj_equi, proj_inv, b, qt, (kt + 1) * 16);
#if HAVE_ASYNC
    if (kt + 1 < 16) asm volatile("s_wait_asynccnt 0xb" ::: "memory");
    else             asm volatile("s_wait_asynccnt 0x0" ::: "memory");
#endif
    __syncthreads();                       // tile kt visible to all waves

    const uint16_t* KM  = (const uint16_t*)(tb + T_KM);
    const float*    EQK = (const float*)(tb + T_EQK);
    const float*    PE  = (const float*)(tb + T_PE);
    const float*    PI  = (const float*)(tb + T_PI);
    const int*      ADJ = (const int*)(tb + T_ADJ);

    // ---- build A fragments for MLP1 (rows = 16 k's of this wave's q) ----
    union Frag { uint32_t u[8]; bf16x16 v; };
    Frag afr[6];
#pragma unroll
    for (int kc = 0; kc < 6; ++kc) {
      const int kbase = kc * 32 + lhal * 8;
#pragma unroll
      for (int v = 0; v < 8; ++v) {
        const int i0 = kbase + ((v < 4) ? 2 * v : 16 + 2 * (v - 4));
        uint16_t lo = getfeat(QM, KM, EQQ, EQK, ql, lrow, i0);
        uint16_t hi = getfeat(QM, KM, EQQ, EQK, ql, lrow, i0 + 1);
        afr[kc].u[v] = pack2(lo, hi);
      }
    }

    // ---- MLP1: 16 N-tiles, K=192 ; silu -> bf16 staged in LDS ----
    for (int j = 0; j < 16; ++j) {
      f32x8 acc = {0.f, 0.f, 0.f, 0.f, 0.f, 0.f, 0.f, 0.f};
#pragma unroll
      for (int kc = 0; kc < 6; ++kc) {
        Frag bfr;
        const uint4* src = (const uint4*)(W1 + ((j * 6 + kc) * 32 + lane) * 8);
        *((uint4*)&bfr.u[0]) = src[0];
        *((uint4*)&bfr.u[4]) = src[1];
        acc = __builtin_amdgcn_wmma_f32_16x16x32_bf16(
            false, afr[kc].v, false, bfr.v, (short)0, acc, false, false);
      }
      const float bias = B1[j * 16 + lrow];
#pragma unroll
      for (int r = 0; r < 8; ++r) {
        float x = acc[r] + bias;
        float s = x * __builtin_amdgcn_rcpf(1.0f + __expf(-x));  // silu, v_rcp
        hidw[(r + lhal * 8) * 256 + j * 16 + lrow] = f2bf(s);
      }
    }
    asm volatile("s_wait_dscnt 0" ::: "memory");   // in-wave LDS transpose

    // ---- MLP2: 5 N-tiles (80 ch), K=256 ----
    f32x8 macc[5];
#pragma unroll
    for (int j2 = 0; j2 < 5; ++j2)
      macc[j2] = (f32x8){0.f, 0.f, 0.f, 0.f, 0.f, 0.f, 0.f, 0.f};
    for (int kc2 = 0; kc2 < 8; ++kc2) {
      Frag af2;
      const int kbase = kc2 * 32 + lhal * 8;
#pragma unroll
      for (int v = 0; v < 8; ++v) {
        const int c0 = kbase + ((v < 4) ? 2 * v : 16 + 2 * (v - 4));
        af2.u[v] = hid32[lrow * 128 + (c0 >> 1)];
      }
#pragma unroll
      for (int j2 = 0; j2 < 5; ++j2) {
        Frag bfr;
        const uint4* src = (const uint4*)(W2 + ((j2 * 8 + kc2) * 32 + lane) * 8);
        *((uint4*)&bfr.u[0]) = src[0];
        *((uint4*)&bfr.u[4]) = src[1];
        macc[j2] = __builtin_amdgcn_wmma_f32_16x16x32_bf16(
            false, af2.v, false, bfr.v, (short)0, macc[j2], false, false);
      }
    }

    // ---- online softmax update (k along VGPR rows; channel along lanes) ----
    float maskv[8];
#pragma unroll
    for (int r = 0; r < 8; ++r)
      maskv[r] = (ADJ[ql * 16 + r + lhal * 8] > 0) ? 0.f : NEGC;

#pragma unroll
    for (int j2 = 0; j2 < 5; ++j2) {
      const float bias2 = B2[j2 * 16 + lrow];
      float msg[8];
      float tmax = -3.0e38f;
#pragma unroll
      for (int r = 0; r < 8; ++r) {
        msg[r] = macc[j2][r] + bias2 + maskv[r];
        tmax = fmaxf(tmax, msg[r]);
      }
      tmax = fmaxf(tmax, __shfl_xor(tmax, 16, 32));
      const float mold = m_[j2];
      const float mnew = fmaxf(mold, tmax);
      const float scale = __expf(mold - mnew);
      m_[j2] = mnew;
      float p[8], ps = 0.f, ps2 = 0.f;
#pragma unroll
      for (int r = 0; r < 8; ++r) {
        p[r] = __expf(msg[r] - mnew);
        ps += p[r];
        ps2 += p[r] * p[r];
      }
      S_[j2]  = S_[j2] * scale + ps;
      S2_[j2] = S2_[j2] * (scale * scale) + ps2;
      if (j2 < 4) {
        const int d = j2 * 16 + lrow;
#pragma unroll
        for (int c = 0; c < 3; ++c) {
          float a = oe[j2][c] * scale;
#pragma unroll
          for (int r = 0; r < 8; ++r)
            a += p[r] * PE[(r + lhal * 8) * 192 + c * 64 + d];
          oe[j2][c] = a;
        }
      } else if (lrow < 8) {
#pragma unroll
        for (int d4 = 0; d4 < 8; ++d4) {
          float4 a = oi[d4];
          a.x *= scale; a.y *= scale; a.z *= scale; a.w *= scale;
#pragma unroll
          for (int r = 0; r < 8; ++r) {
            const float4 pv =
                *((const float4*)&PI[(r + lhal * 8) * 256 + lrow * 32 + d4 * 4]);
            a.x += p[r] * pv.x; a.y += p[r] * pv.y;
            a.z += p[r] * pv.z; a.w += p[r] * pv.w;
          }
          oi[d4] = a;
        }
      }
    }
  } // kt

  // ---- finalize: combine lane halves, apply sqrt(S2)/S^2, output GEMMs ----
  float* TE = (float*)hidw;
  float* VI = ((float*)hidw) + 192;
  const int qg = qt * 8 + ql;

#pragma unroll
  for (int j2 = 0; j2 < 4; ++j2) {
    const float St  = S_[j2]  + __shfl_xor(S_[j2], 16, 32);
    const float S2t = S2_[j2] + __shfl_xor(S2_[j2], 16, 32);
    const float f = __fsqrt_rn(S2t) / (St * St);
#pragma unroll
    for (int c = 0; c < 3; ++c) {
      const float ot = oe[j2][c] + __shfl_xor(oe[j2][c], 16, 32);
      if (lane < 16) TE[c * 64 + j2 * 16 + lrow] = ot * f;
    }
  }
  {
    const float St  = S_[4]  + __shfl_xor(S_[4], 16, 32);
    const float S2t = S2_[4] + __shfl_xor(S2_[4], 16, 32);
    const float f = __fsqrt_rn(S2t) / (St * St);
#pragma unroll
    for (int d4 = 0; d4 < 8; ++d4) {
      float4 a = oi[d4];
      a.x += __shfl_xor(a.x, 16, 32); a.y += __shfl_xor(a.y, 16, 32);
      a.z += __shfl_xor(a.z, 16, 32); a.w += __shfl_xor(a.w, 16, 32);
      if (lane < 8) {
        const int base = lane * 32 + d4 * 4;
        VI[base + 0] = a.x * f; VI[base + 1] = a.y * f;
        VI[base + 2] = a.z * f; VI[base + 3] = a.w * f;
      }
    }
  }
  asm volatile("s_wait_dscnt 0" ::: "memory");

  float* outp = out + ((size_t)(b * 256) + qg) * 448;
#pragma unroll
  for (int it = 0; it < 6; ++it) {
    const int i = lane + it * 32;
    const int c = i >> 6, dp = i & 63;
    float acc = 0.f;
    for (int d = 0; d < 64; ++d) acc += TE[c * 64 + d] * w_equi_out[d * 64 + dp];
    outp[i] = acc;
  }
#pragma unroll
  for (int it = 0; it < 8; ++it) {
    const int dp = lane + it * 32;
    float acc = b_inv_out[dp];
    for (int j = 0; j < 256; ++j) acc += VI[j] * w_inv_out[j * 256 + dp];
    outp[192 + dp] = acc;
  }
}

// -------------------------- launcher ---------------------------------------

extern "C" void kernel_launch(void* const* d_in, const int* in_sizes, int n_in,
                              void* d_out, int out_size, void* d_ws,
                              size_t ws_size, hipStream_t stream) {
  (void)in_sizes; (void)n_in; (void)out_size; (void)ws_size;
  const float* equis = (const float*)d_in[0];
  const float* invs  = (const float*)d_in[1];
  const int*   adj   = (const int*)d_in[2];
  const float* w_q   = (const float*)d_in[3];
  const float* b_q   = (const float*)d_in[4];
  const float* w_k   = (const float*)d_in[5];
  const float* b_k   = (const float*)d_in[6];
  const float* w1    = (const float*)d_in[7];
  const float* b1    = (const float*)d_in[8];
  const float* w2    = (const float*)d_in[9];
  const float* b2    = (const float*)d_in[10];
  const float* wco   = (const float*)d_in[11];
  const float* weo   = (const float*)d_in[12];
  const float* wii   = (const float*)d_in[13];
  const float* bii   = (const float*)d_in[14];
  const float* wio   = (const float*)d_in[15];
  const float* bio   = (const float*)d_in[16];
  float* outp = (float*)d_out;
  char* ws = (char*)d_ws;

  uint32_t* w1p = (uint32_t*)(ws + WS_W1);
  uint32_t* w2p = (uint32_t*)(ws + WS_W2);
  uint16_t* qmb = (uint16_t*)(ws + WS_QM);
  uint16_t* kmb = (uint16_t*)(ws + WS_KM);
  float* pe = (float*)(ws + WS_PE);
  float* pi = (float*)(ws + WS_PI);

  (void)hipFuncSetAttribute((const void*)semla_main,
                            hipFuncAttributeMaxDynamicSharedMemorySize,
                            (int)SMEM_BYTES);

  prep_nodes<<<2048, 256, 0, stream>>>(invs, w_q, b_q, w_k, b_k, wii, bii,
                                       qmb, kmb, pi);
  prep_equi<<<6144, 64, 0, stream>>>(equis, wco, pe);
  prep_pack<<<136, 256, 0, stream>>>(w1, w2, w1p, w2p);
  semla_main<<<256, 256, SMEM_BYTES, stream>>>(equis, adj, b1, b2, weo, wio,
                                               bio, qmb, kmb, pe, pi, w1p, w2p,
                                               outp);
}
